// TransducerJoiner_29824252903691
// MI455X (gfx1250) — matrix operand order repaired and checked by reading, not compiled, not run on hardware
//
#include <hip/hip_runtime.h>
#include <hip/hip_bf16.h>
#include <math.h>

// ---------------------------------------------------------------------------
// Transducer joiner, fused for gfx1250 (CDNA5, wave32, WMMA bf16).
//   Stage 1: ps = src @ W1[:D], pt = tgt @ W1[D:]     (fp32, tiny GEMMs)
//   Stage 2 (fused): h = tanh(ps + pt + b1) -> bf16 in LDS (32x640 tile)
//                    logits = h @ W2_bf16 + b2         (v_wmma_f32_16x16x32_bf16)
//                    out = log_softmax(logits)         (in-register + LDS merge)
// M-tile = 32 rows/block (512 thr, 16 waves) halves W2 L2 traffic vs M=16.
// ---------------------------------------------------------------------------

#define B_  4
#define T_  128
#define U_  64
#define D_  512
#define H_  640
#define V_  1024

#define KT_ (H_ / 32)      // 20 k-tiles of 32
#define NT_ (V_ / 16)      // 64 n-tiles of 16

typedef __attribute__((ext_vector_type(16))) __bf16 v16bf;
typedef __attribute__((ext_vector_type(8)))  float  v8f;

union V16U { uint4 u[2]; v16bf v; };

__device__ inline unsigned short f32_to_bf16_rne(float f) {
    unsigned int x = __float_as_uint(f);
    unsigned int r = x + 0x7fffu + ((x >> 16) & 1u);
    return (unsigned short)(r >> 16);
}

// ---------------------------------------------------------------------------
// Stage 1: rows 0..511 -> ps = src @ W1[0:512,:]; rows 512..767 -> pt.
// ---------------------------------------------------------------------------
__global__ void joiner_proj_kernel(const float* __restrict__ src,
                                   const float* __restrict__ tgt,
                                   const float* __restrict__ W1,
                                   float* __restrict__ ps,   // [512][640]
                                   float* __restrict__ pt)   // [256][640]
{
    int h = blockIdx.x * blockDim.x + threadIdx.x;
    if (h >= H_) return;
    int row = blockIdx.y;

    const float* enc;
    int koff;
    float* outp;
    if (row < B_ * T_) { enc = src + (size_t)row * D_;             koff = 0;  outp = ps + (size_t)row * H_; }
    else               { enc = tgt + (size_t)(row - B_*T_) * D_;   koff = D_; outp = pt + (size_t)(row - B_*T_) * H_; }

    float acc = 0.f;
    #pragma unroll 4
    for (int k = 0; k < D_; ++k)
        acc = fmaf(enc[k], W1[(size_t)(koff + k) * H_ + h], acc);
    outp[h] = acc;
}

// ---------------------------------------------------------------------------
// W2 (640x1024 f32) -> bf16, pre-swizzled into per-lane B-fragment order:
//   tile (kt,nt): lane L holds e=0..15 contiguous, elem = W2[kt*32+Kb+e][nt*16+L%16]
//   where Kb = (L<16 ? 0 : 16).  One 32B read per lane per fragment at use time.
// ---------------------------------------------------------------------------
__global__ void w2_swizzle_kernel(const float* __restrict__ W2,
                                  unsigned short* __restrict__ w2t)
{
    int idx = blockIdx.x * blockDim.x + threadIdx.x;
    if (idx >= KT_ * NT_ * 32 * 16) return;
    int e    =  idx        & 15;
    int lane = (idx >> 4)  & 31;
    int nt   = (idx >> 9)  & 63;
    int kt   =  idx >> 15;
    int kb   = (lane < 16) ? 0 : 16;
    int k = kt * 32 + kb + e;
    int n = nt * 16 + (lane & 15);
    w2t[idx] = f32_to_bf16_rne(W2[(size_t)k * V_ + n]);
}

// ---------------------------------------------------------------------------
// Main fused kernel. Block = 32 rows (fixed b,t; 32 consecutive u).
// 512 threads = 16 waves; wave w owns columns [w*64, w*64+64) (4 n-tiles)
// and both 16-row m-tiles.  B-fragments held across m-tiles (no extra L2).
// ---------------------------------------------------------------------------
__global__ void __launch_bounds__(512)
joiner_main_kernel(const float* __restrict__ ps,
                   const float* __restrict__ pt,
                   const float* __restrict__ b1,
                   const unsigned short* __restrict__ w2t,
                   const float* __restrict__ b2,
                   float* __restrict__ out)
{
    __shared__ __align__(16) unsigned short hTile[32][648]; // bf16 bits, padded
    __shared__ float sb[H_];          // ps row + b1 (shared by all 32 rows)
    __shared__ float redMax[32][16];
    __shared__ float redSum[32][16];

    const int tid  = threadIdx.x;
    const int w    = tid >> 5;        // wave 0..15
    const int lane = tid & 31;
    const int half = lane >> 4;
    const int l16  = lane & 15;

    // tile -> (b, t, u0)   (32 | U, so fixed b,t per block)
    const int g    = blockIdx.x * 32;
    const int b    = g >> 13;              // / (T*U)
    const int rem  = g & 8191;
    const int t    = rem >> 6;             // / U
    const int u0   = rem & 63;             // 0 or 32

    const float* psRow = ps + (size_t)(b * T_ + t) * H_;
    const float* ptRow = pt + (size_t)(b * U_ + u0) * H_;

    // ---- build bf16 activation tile in LDS ----
    for (int k = tid; k < H_; k += 512) sb[k] = psRow[k] + b1[k];
    __syncthreads();
    for (int idx = tid; idx < 32 * H_; idx += 512) {
        int r = idx / H_;
        int k = idx - r * H_;
        float v = tanhf(sb[k] + ptRow[(size_t)r * H_ + k]);
        hTile[r][k] = f32_to_bf16_rne(v);
    }
    __syncthreads();

    // ---- GEMM: 32x640 (LDS) x 640x64 (this wave's W2 slice) ----
    v8f acc[2][4];
    #pragma unroll
    for (int mt = 0; mt < 2; ++mt)
        #pragma unroll
        for (int j = 0; j < 4; ++j)
            #pragma unroll
            for (int v = 0; v < 8; ++v) acc[mt][j][v] = 0.f;

    for (int kt = 0; kt < KT_; ++kt) {
        // B fragments for this wave's 4 n-tiles (held across both m-tiles)
        V16U bu[4];
        #pragma unroll
        for (int j = 0; j < 4; ++j) {
            const uint4* bp = reinterpret_cast<const uint4*>(
                w2t + ((size_t)(kt * NT_ + (w * 4 + j)) * 32 + lane) * 16);
            bu[j].u[0] = bp[0];
            bu[j].u[1] = bp[1];
        }
        const int ka = kt * 32 + half * 16;
        #pragma unroll
        for (int mt = 0; mt < 2; ++mt) {
            // A fragment: lane L<16: row l16, k = kt*32 + {0..15}
            //             lane L>=16: row l16, k = kt*32 + {16..31}
            V16U au;
            au.u[0] = *reinterpret_cast<const uint4*>(&hTile[mt * 16 + l16][ka]);
            au.u[1] = *reinterpret_cast<const uint4*>(&hTile[mt * 16 + l16][ka + 8]);
            #pragma unroll
            for (int j = 0; j < 4; ++j) {
                acc[mt][j] = __builtin_amdgcn_wmma_f32_16x16x32_bf16(
                    false, au.v, false, bu[j].v, (short)0, acc[mt][j], false, false);
            }
        }
    }

    // ---- + b2 ----
    #pragma unroll
    for (int j = 0; j < 4; ++j) {
        const float bb = b2[(w * 4 + j) * 16 + l16];
        #pragma unroll
        for (int mt = 0; mt < 2; ++mt)
            #pragma unroll
            for (int v = 0; v < 8; ++v) acc[mt][j][v] += bb;
    }

    // ---- per-row log-softmax over V=1024 ----
    // acc[mt][j][v] holds row m = mt*16 + v + 8*half, col n = (w*4+j)*16 + l16.
    #pragma unroll
    for (int mt = 0; mt < 2; ++mt)
        #pragma unroll
        for (int v = 0; v < 8; ++v) {
            float mx = -INFINITY;
            #pragma unroll
            for (int j = 0; j < 4; ++j) mx = fmaxf(mx, acc[mt][j][v]);
            #pragma unroll
            for (int off = 8; off >= 1; off >>= 1)   // stays within 16-lane half
                mx = fmaxf(mx, __shfl_xor(mx, off, 32));
            float s = 0.f;
            #pragma unroll
            for (int j = 0; j < 4; ++j) s += __expf(acc[mt][j][v] - mx);
            #pragma unroll
            for (int off = 8; off >= 1; off >>= 1)
                s += __shfl_xor(s, off, 32);
            if (l16 == 0) {
                redMax[mt * 16 + v + 8 * half][w] = mx;
                redSum[mt * 16 + v + 8 * half][w] = s;
            }
        }
    __syncthreads();

    float logZ[2][8];
    #pragma unroll
    for (int mt = 0; mt < 2; ++mt)
        #pragma unroll
        for (int v = 0; v < 8; ++v) {
            const int m = mt * 16 + v + 8 * half;
            float gm = -INFINITY;
            #pragma unroll
            for (int ww = 0; ww < 16; ++ww) gm = fmaxf(gm, redMax[m][ww]);
            float gs = 0.f;
            #pragma unroll
            for (int ww = 0; ww < 16; ++ww)
                gs += __expf(redMax[m][ww] - gm) * redSum[m][ww];
            logZ[mt][v] = gm + logf(gs);
        }

    // ---- store ----
    float* outBase = out + (size_t)blockIdx.x * 32 * V_;
    #pragma unroll
    for (int mt = 0; mt < 2; ++mt)
        #pragma unroll
        for (int j = 0; j < 4; ++j) {
            const int n = (w * 4 + j) * 16 + l16;
            #pragma unroll
            for (int v = 0; v < 8; ++v) {
                const int m = mt * 16 + v + 8 * half;
                outBase[(size_t)m * V_ + n] = acc[mt][j][v] - logZ[mt][v];
            }
        }
}

// Copy the two int32 length vectors (tuple outputs 2 & 3) into d_out tail.
__global__ void joiner_tail_kernel(const int* __restrict__ sl,
                                   const int* __restrict__ tl,
                                   int* __restrict__ tail)
{
    int i = threadIdx.x;
    if (i < 4)      tail[i] = sl[i];
    else if (i < 8) tail[i] = tl[i - 4];
}

extern "C" void kernel_launch(void* const* d_in, const int* in_sizes, int n_in,
                              void* d_out, int out_size, void* d_ws, size_t ws_size,
                              hipStream_t stream)
{
    (void)in_sizes; (void)n_in; (void)out_size; (void)ws_size;

    const float* src = (const float*)d_in[0];           // (4,128,512)
    const int*   sl  = (const int*)  d_in[1];           // (4,)
    const float* tgt = (const float*)d_in[2];           // (4,64,512)
    const int*   tl  = (const int*)  d_in[3];           // (4,)
    const float* W1  = (const float*)d_in[4];           // (1024,640)
    const float* b1  = (const float*)d_in[5];           // (640,)
    const float* W2  = (const float*)d_in[6];           // (640,1024)
    const float* b2  = (const float*)d_in[7];           // (1024,)

    char* ws = (char*)d_ws;
    float*          ps  = (float*)ws;                                   // 512*640 f32
    float*          pt  = (float*)(ws + (size_t)B_*T_*H_*4);            // 256*640 f32
    unsigned short* w2t = (unsigned short*)(ws + (size_t)(B_*T_ + B_*U_)*H_*4); // 640*1024 bf16

    float* out = (float*)d_out;

    // Stage 1: projections (768 rows x 640 cols)
    joiner_proj_kernel<<<dim3((H_ + 255) / 256, B_*T_ + B_*U_), 256, 0, stream>>>(
        src, tgt, W1, ps, pt);

    // W2 -> bf16, fragment-swizzled
    w2_swizzle_kernel<<<(KT_ * NT_ * 32 * 16 + 255) / 256, 256, 0, stream>>>(W2, w2t);

    // Fused tanh + GEMM (WMMA bf16) + log_softmax, 32 rows/block
    joiner_main_kernel<<<(B_ * T_ * U_) / 32, 512, 0, stream>>>(
        ps, pt, b1, w2t, b2, out);

    // Tuple tail: lengths
    joiner_tail_kernel<<<1, 8, 0, stream>>>(sl, tl, (int*)d_out + (size_t)B_*T_*U_*V_);
}
